// PointMeta_10101763080207
// MI455X (gfx1250) — matrix operand order repaired and checked by reading, not compiled, not run on hardware
//
#include <hip/hip_runtime.h>

typedef __attribute__((ext_vector_type(16))) _Float16 v16h;
typedef __attribute__((ext_vector_type(8)))  float    v8f;
typedef unsigned int u32;
typedef __attribute__((ext_vector_type(4))) u32 v4u;
typedef __attribute__((ext_vector_type(8))) int v8i;
typedef __attribute__((ext_vector_type(4))) int v4i;

// ---------------------------------------------------------------------
// Tensor Data Mover: async 2D tile (rows x cols of f32) global -> LDS.
// Descriptor per CDNA5 ISA (D# group0/group1).
// This toolchain exposes the 6-arg builtin (clang-23 / therock form):
//   (uint32x4 g0, int32x8 g1, int32x4 g2, int32x4 g3, int32x8 g4, i32 cpol)
// ---------------------------------------------------------------------
static __device__ __forceinline__ void tdm_load_2d(u32 lds_addr, const float* gptr,
                                                   u32 tdim0, u32 tdim1,
                                                   u32 tile0, u32 tile1, u32 stride0) {
  unsigned long long ga = (unsigned long long)(uintptr_t)gptr;
  v4u g0;
  g0[0] = 1u;                                   // count=1, user descriptor, no gather
  g0[1] = lds_addr;                             // LDS byte address
  g0[2] = (u32)ga;                              // global_addr[31:0]
  g0[3] = (u32)(ga >> 32) | (2u << 30);         // global_addr[56:32] | type=2 (image)
  v8i g1;
  g1[0] = (int)(2u << 16);                      // workgroup_mask=0, data_size=2 (4B)
  g1[1] = (int)((tdim0 & 0xFFFFu) << 16);       // tensor_dim0[15:0]
  g1[2] = (int)((tdim0 >> 16) | ((tdim1 & 0xFFFFu) << 16));
  g1[3] = (int)((tdim1 >> 16) | (tile0 << 16)); // tile_dim0
  g1[4] = (int)(tile1 & 0xFFFFu);               // tile_dim1 (tile_dim2=0)
  g1[5] = (int)stride0;                         // tensor_dim0_stride[31:0]
  g1[6] = 0;                                    // stride0[47:32], stride1[15:0]
  g1[7] = 0;
  v4i z4 = {0, 0, 0, 0};
  v8i z8 = {0, 0, 0, 0, 0, 0, 0, 0};
  __builtin_amdgcn_tensor_load_to_lds(g0, g1, z4, z4, z8, 0);
}

// =====================================================================
// WMMA GEMM:  Y[b,j,o] = relu( g[o]*(sum_c X[b,j,c]*W[o,c]) + beta[o] (+R) )
//   X: (B,n,Cin) point-major, W: (Cout,Cin), Y/R: (B,n,Cout)
//   n%16==0, Cin%32==0, Cout%64==0.
// Block = 4 waves = one 16-point x 64-out-channel macro tile.
// Data path: TDM(tensor_load_to_lds, double buffered) -> ds_load -> cvt -> WMMA.
// A layout (16x32 f16): lane = M row, elem e -> K = (h<<4)+e
// B layout (32x16 f16): lane = N col, elem e -> K = (h<<4)+e
// C/D (16x16 f32):      lane = N col, vgpr r -> M = r + 8h
// =====================================================================
template <bool HAS_R>
__global__ void __launch_bounds__(128) k_gemm_bn_relu(
    const float* __restrict__ X, const float* __restrict__ W,
    const float* __restrict__ g, const float* __restrict__ bet,
    const float* __restrict__ R, float* __restrict__ Y, int n, int Cin, int Cout) {
  __shared__ float sA[2][16 * 32];   // 4 KB : 16 points x 32 channels, ping-pong
  __shared__ float sB[2][64 * 32];   // 16 KB: 64 outch  x 32 channels, ping-pong

  const int w = threadIdx.y;         // wave 0..3
  const int lane = threadIdx.x;
  const int h = lane >> 4, ln = lane & 15;
  const int nG = Cout >> 6;
  const int mT = blockIdx.x / nG;
  const int gN = blockIdx.x - mT * nG;
  const int b = blockIdx.y;
  const int j0 = mT << 4;
  const int o0g = gN << 6;
  const int o0 = o0g + (w << 4);

  const float* Abase = X + ((size_t)b * n + j0) * Cin;
  const float* Bbase = W + (size_t)o0g * Cin;
  const u32 ldsA[2] = {(u32)(uintptr_t)&sA[0][0], (u32)(uintptr_t)&sA[1][0]};
  const u32 ldsB[2] = {(u32)(uintptr_t)&sB[0][0], (u32)(uintptr_t)&sB[1][0]};

  const int steps = Cin >> 5;
  if (w == 0) {  // wave 0 drives the TDM pipeline (EXEC ignored by tensor ops)
    tdm_load_2d(ldsA[0], Abase, (u32)Cin, 1u << 20, 32u, 16u, (u32)Cin);
    tdm_load_2d(ldsB[0], Bbase, (u32)Cin, 1u << 20, 32u, 64u, (u32)Cin);
  }

  v8f acc = {0.f, 0.f, 0.f, 0.f, 0.f, 0.f, 0.f, 0.f};
  for (int s = 0; s < steps; ++s) {
    const int cur = s & 1;
    if (w == 0) {
      if (s + 1 < steps) {           // prefetch next K-slab, then wait for current
        const int nxt = (s + 1) & 1;
        const int c0n = (s + 1) << 5;
        tdm_load_2d(ldsA[nxt], Abase + c0n, (u32)Cin, 1u << 20, 32u, 16u, (u32)Cin);
        tdm_load_2d(ldsB[nxt], Bbase + c0n, (u32)Cin, 1u << 20, 32u, 64u, (u32)Cin);
        __builtin_amdgcn_s_wait_tensorcnt(2);  // in-order: current pair landed
      } else {
        __builtin_amdgcn_s_wait_tensorcnt(0);
      }
    }
    __syncthreads();                 // LDS tiles for step s visible to all waves

    const float* ar = &sA[cur][ln * 32 + (h << 4)];
    const float* br = &sB[cur][((w << 4) + ln) * 32 + (h << 4)];
    v16h a, bm;
#pragma unroll
    for (int e = 0; e < 8; ++e) a[e] = (_Float16)ar[e];
#pragma unroll
    for (int e = 0; e < 8; ++e) a[e + 8] = (_Float16)ar[e + 8];
#pragma unroll
    for (int e = 0; e < 16; ++e) bm[e] = (_Float16)br[e];
    acc = __builtin_amdgcn_wmma_f32_16x16x32_f16(false, a, false, bm,
                                                 (short)0, acc, false, false);
    __syncthreads();                 // done reading before buffer is re-filled
  }

  const int o = o0 + ln;
  const float gg = g[o], bb = bet[o];
  float* Yb = Y + (size_t)b * n * Cout;
  const float* Rb = HAS_R ? (R + (size_t)b * n * Cout) : nullptr;
#pragma unroll
  for (int r = 0; r < 8; ++r) {
    int j = j0 + r + (h << 3);
    float z = gg * acc[r] + bb;
    if (HAS_R) z += Rb[(size_t)j * Cout + o];
    Yb[(size_t)j * Cout + o] = fmaxf(z, 0.0f);
  }
}

// ---------------- input embedding: x64 = [pos_z, x0, x1, x2] @ inW + inb ----
__global__ void k_embed(const float* __restrict__ pos, const float* __restrict__ x,
                        const float* __restrict__ W, const float* __restrict__ b,
                        float* __restrict__ out, int n, int total) {
  int gid = blockIdx.x * blockDim.x + threadIdx.x;
  if (gid >= total) return;
  int o = gid & 63;
  int j = (gid >> 6) % n;
  int bb = gid / (n << 6);
  size_t p = ((size_t)bb * n + j) * 3;
  float v = b[o] + pos[p + 2] * W[o] + x[p] * W[64 + o] + x[p + 1] * W[128 + o] +
            x[p + 2] * W[192 + o];
  out[((size_t)bb * n + j) * 64 + o] = v;
}

// ---------------- farthest point sampling (matches lax.scan semantics) ------
__global__ void k_fps(const float* __restrict__ P, int n, int m, int* __restrict__ out) {
  __shared__ float dis[8192];
  __shared__ float rv[256];
  __shared__ int ri[256];
  __shared__ int s_last;
  const int b = blockIdx.x;
  const float* pb = P + (size_t)b * n * 3;
  const int t = threadIdx.x;
  for (int j = t; j < n; j += 256) dis[j] = 1e10f;
  if (t == 0) { out[(size_t)b * m] = 0; s_last = 0; }
  __syncthreads();
  for (int it = 1; it < m; ++it) {
    const int last = s_last;
    const float lx = pb[last * 3], ly = pb[last * 3 + 1], lz = pb[last * 3 + 2];
    float bestv = -1.0f; int besti = 0x7fffffff;
    for (int j = t; j < n; j += 256) {
      float dx = pb[j * 3] - lx, dy = pb[j * 3 + 1] - ly, dz = pb[j * 3 + 2] - lz;
      float d = dx * dx + dy * dy + dz * dz;
      float dd = fminf(dis[j], d);
      dis[j] = dd;
      if (dd > bestv) { bestv = dd; besti = j; }  // strict >: first max in stride-set
    }
    rv[t] = bestv; ri[t] = besti;
    __syncthreads();
    for (int s = 128; s > 0; s >>= 1) {
      if (t < s) {
        if (rv[t + s] > rv[t] || (rv[t + s] == rv[t] && ri[t + s] < ri[t])) {
          rv[t] = rv[t + s]; ri[t] = ri[t + s];
        }
      }
      __syncthreads();
    }
    if (t == 0) { s_last = ri[0]; out[(size_t)b * m + it] = ri[0]; }
    __syncthreads();
  }
}

// ---------------- gather xyz by index ---------------------------------------
__global__ void k_gather3(const float* __restrict__ P, const int* __restrict__ idx,
                          float* __restrict__ out, int n, int m, int total) {
  int gid = blockIdx.x * blockDim.x + threadIdx.x;
  if (gid >= total) return;
  int j = gid % m, b = gid / m;
  int i = idx[(size_t)b * m + j];
  const float* p = P + ((size_t)b * n + i) * 3;
  float* o = out + ((size_t)b * m + j) * 3;
  o[0] = p[0]; o[1] = p[1]; o[2] = p[2];
}

// ---------------- ball query: first K indices with d2 < r^2, pad with first -
__global__ void k_ball(const float* __restrict__ C, int m, const float* __restrict__ P,
                       int n, float rr, int* __restrict__ idx, int total) {
  int gid = blockIdx.x * blockDim.x + threadIdx.x;
  if (gid >= total) return;
  int j = gid % m, b = gid / m;
  const float* c = C + ((size_t)b * m + j) * 3;
  float cx = c[0], cy = c[1], cz = c[2];
  const float* p = P + (size_t)b * n * 3;
  int* o = idx + (size_t)gid * 32;
  int cnt = 0, first = 0;
  for (int i = 0; i < n; ++i) {
    float dx = p[i * 3] - cx, dy = p[i * 3 + 1] - cy, dz = p[i * 3 + 2] - cz;
    if (dx * dx + dy * dy + dz * dz < rr) {
      if (cnt == 0) first = i;
      o[cnt++] = i;
      if (cnt == 32) break;
    }
  }
  for (; cnt < 32; ++cnt) o[cnt] = first;
}

// ---- grouped max of (gathered feature + relu(bn(posconv(rel/r)))) ----------
//   out[b,j,o] = max_k F[b,idx,o] + relu(g*(W3[o].(SP[idx]-CP[j])/r)+beta)
__global__ void k_groupmax(const int* __restrict__ idx, const float* __restrict__ SP,
                           int nsrc, const float* __restrict__ CP,
                           const float* __restrict__ F, const float* __restrict__ W3,
                           const float* __restrict__ g, const float* __restrict__ bet,
                           float inv_r, float* __restrict__ out, int m, int C, int total) {
  int gid = blockIdx.x * blockDim.x + threadIdx.x;
  if (gid >= total) return;
  int o = gid % C;
  int j = (gid / C) % m;
  int b = gid / (C * m);
  const float* sp = SP + (size_t)b * nsrc * 3;
  const float* cp = CP + ((size_t)b * m + j) * 3;
  const float cx = cp[0], cy = cp[1], cz = cp[2];
  const float w0 = W3[o * 3] * inv_r, w1 = W3[o * 3 + 1] * inv_r, w2 = W3[o * 3 + 2] * inv_r;
  const float gg = g[o], bb = bet[o];
  const int* id = idx + ((size_t)b * m + j) * 32;
  const float* fb = F + (size_t)b * nsrc * C;
  float best = -3.4e38f;
#pragma unroll 4
  for (int k = 0; k < 32; ++k) {
    int i = id[k];
    float pe = fmaxf(gg * (w0 * (sp[i * 3] - cx) + w1 * (sp[i * 3 + 1] - cy) +
                           w2 * (sp[i * 3 + 2] - cz)) + bb, 0.f);
    best = fmaxf(best, fb[(size_t)i * C + o] + pe);
  }
  out[((size_t)b * m + j) * C + o] = best;
}

// ---------------- 3-NN (squared dist) + inverse-distance weights -------------
__global__ void k_knn3(const float* __restrict__ P1, int n1, const float* __restrict__ P2,
                       int n2, int* __restrict__ wi, float* __restrict__ ww, int total) {
  int gid = blockIdx.x * blockDim.x + threadIdx.x;
  if (gid >= total) return;
  int j = gid % n1, b = gid / n1;
  const float* p = P1 + ((size_t)b * n1 + j) * 3;
  const float px = p[0], py = p[1], pz = p[2];
  const float* q = P2 + (size_t)b * n2 * 3;
  float d0 = 3.4e38f, d1 = 3.4e38f, d2 = 3.4e38f;
  int i0 = 0, i1 = 0, i2 = 0;
  for (int i = 0; i < n2; ++i) {
    float dx = q[i * 3] - px, dy = q[i * 3 + 1] - py, dz = q[i * 3 + 2] - pz;
    float d = dx * dx + dy * dy + dz * dz;
    if (d < d0)      { d2 = d1; i2 = i1; d1 = d0; i1 = i0; d0 = d; i0 = i; }
    else if (d < d1) { d2 = d1; i2 = i1; d1 = d; i1 = i; }
    else if (d < d2) { d2 = d; i2 = i; }
  }
  float w0 = 1.f / (d0 + 1e-8f), w1 = 1.f / (d1 + 1e-8f), w2 = 1.f / (d2 + 1e-8f);
  float s = w0 + w1 + w2;
  size_t o = (size_t)gid * 3;
  wi[o] = i0; wi[o + 1] = i1; wi[o + 2] = i2;
  ww[o] = w0 / s; ww[o + 1] = w1 / s; ww[o + 2] = w2 / s;
}

// ------- interpolate coarse features + concat fine features -----------------
__global__ void k_interp_cat(const float* __restrict__ F2, int c2,
                             const float* __restrict__ F1, int c1,
                             const int* __restrict__ wi, const float* __restrict__ ww,
                             float* __restrict__ out, int n1, int n2, int total) {
  int gid = blockIdx.x * blockDim.x + threadIdx.x;
  if (gid >= total) return;
  const int C = c1 + c2;
  int c = gid % C;
  int j = (gid / C) % n1;
  int b = gid / (C * n1);
  float v;
  if (c < c2) {
    size_t o = ((size_t)b * n1 + j) * 3;
    const float* fb = F2 + (size_t)b * n2 * c2;
    v = ww[o] * fb[(size_t)wi[o] * c2 + c] + ww[o + 1] * fb[(size_t)wi[o + 1] * c2 + c] +
        ww[o + 2] * fb[(size_t)wi[o + 2] * c2 + c];
  } else {
    v = F1[((size_t)b * n1 + j) * c1 + (c - c2)];
  }
  out[((size_t)b * n1 + j) * C + c] = v;
}

// ---------------- head: out[b,cls,j] = b2[cls] + h[b,j,:]·W2[cls,:] ----------
__global__ void k_head(const float* __restrict__ H, const float* __restrict__ W2,
                       const float* __restrict__ b2, float* __restrict__ out, int n,
                       int total) {
  int gid = blockIdx.x * blockDim.x + threadIdx.x;
  if (gid >= total) return;
  int cls = gid % 13;
  int j = (gid / 13) % n;
  int b = gid / (13 * n);
  const float* hr = H + ((size_t)b * n + j) * 64;
  const float* wr = W2 + cls * 64;
  float s = b2[cls];
#pragma unroll 8
  for (int c = 0; c < 64; ++c) s += hr[c] * wr[c];
  out[((size_t)b * 13 + cls) * n + j] = s;
}

// =====================================================================
static void gemm(const float* X, const float* W, const float* g, const float* b,
                 const float* R, float* Y, int Bn, int n, int Cin, int Cout,
                 hipStream_t s) {
  dim3 grid((n >> 4) * (Cout >> 6), Bn);
  dim3 blk(32, 4);
  if (R)
    k_gemm_bn_relu<true><<<grid, blk, 0, s>>>(X, W, g, b, R, Y, n, Cin, Cout);
  else
    k_gemm_bn_relu<false><<<grid, blk, 0, s>>>(X, W, g, b, R, Y, n, Cin, Cout);
}

extern "C" void kernel_launch(void* const* d_in, const int* in_sizes, int n_in,
                              void* d_out, int out_size, void* d_ws, size_t ws_size,
                              hipStream_t stream) {
  (void)in_sizes; (void)n_in; (void)out_size; (void)ws_size;
  const int B = 2, N = 8192, KG = 32;
  const float* pos = (const float*)d_in[0];
  const float* xin = (const float*)d_in[1];
  const float* inW = (const float*)d_in[2];
  const float* inb = (const float*)d_in[3];
  // stage s param k (insertion order): 0 sa_x, 1/2 sa_x_bn g/b, 3 sa_pos, 4/5 bn,
  // 6 pe_pos, 7/8 bn, 9 inv.c1, 10/11 bn1, 12 inv.c21, 13/14 bn21, 15 inv.c22, 16/17 bn22
  auto SPm = [&](int s, int k) { return (const float*)d_in[4 + 18 * s + k]; };
  auto FPm = [&](int l, int k) { return (const float*)d_in[76 + 6 * l + k]; };
  const float* out_c1 = (const float*)d_in[100];
  const float* out_g  = (const float*)d_in[101];
  const float* out_b  = (const float*)d_in[102];
  const float* out_c2 = (const float*)d_in[103];
  const float* out_b2 = (const float*)d_in[104];

  char* base = (char*)d_ws;
  size_t off = 0;
  auto aalloc = [&](size_t bytes) -> void* {
    size_t o = (off + 255) & ~(size_t)255;
    off = o + bytes;
    return (void*)(base + o);
  };

  const int ns[5] = {8192, 2048, 512, 128, 32};
  const int cs[5] = {64, 128, 256, 512, 1024};
  const float rad[4] = {0.1f, 0.2f, 0.4f, 0.8f};

  const float* Plv[5];
  const float* Flv[5];

  // ---- input embedding -> f0 (B,N,64)
  float* f0 = (float*)aalloc((size_t)B * N * 64 * 4);
  {
    int tot = B * N * 64;
    k_embed<<<(tot + 255) / 256, 256, 0, stream>>>(pos, xin, inW, inb, f0, N, tot);
  }
  Plv[0] = pos; Flv[0] = f0;

  // ---- 4 set-abstraction stages (each followed by one inverted-residual block)
  for (int s = 0; s < 4; ++s) {
    const int n = ns[s], m = ns[s + 1], ci = cs[s], co = cs[s + 1];
    const float r = rad[s], r2 = 2.f * r;
    const float* P = Plv[s];
    const float* F = Flv[s];

    float* cent = (float*)aalloc((size_t)B * m * 3 * 4);   // persistent
    float* fout = (float*)aalloc((size_t)B * m * co * 4);  // persistent
    size_t mark = off;                                     // transient region

    int* fpsidx = (int*)aalloc((size_t)B * m * 4);
    k_fps<<<B, 256, 0, stream>>>(P, n, m, fpsidx);
    {
      int tot = B * m;
      k_gather3<<<(tot + 255) / 256, 256, 0, stream>>>(P, fpsidx, cent, n, m, tot);
    }

    // f = relu(bn(sa_x . feat))  over all n input points
    float* fconv = (float*)aalloc((size_t)B * n * co * 4);
    gemm(F, SPm(s, 0), SPm(s, 1), SPm(s, 2), nullptr, fconv, B, n, ci, co, stream);

    // sa ball query + grouped max with positional encoding (sa_pos, radius r)
    int* idx_sa = (int*)aalloc((size_t)B * m * KG * 4);
    {
      int tot = B * m;
      k_ball<<<(tot + 63) / 64, 64, 0, stream>>>(cent, m, P, n, r * r, idx_sa, tot);
    }
    float* fstage = (float*)aalloc((size_t)B * m * co * 4);
    {
      int tot = B * m * co;
      k_groupmax<<<(tot + 255) / 256, 256, 0, stream>>>(
          idx_sa, P, n, cent, fconv, SPm(s, 3), SPm(s, 4), SPm(s, 5), 1.f / r,
          fstage, m, co, tot);
    }

    // pe ball query on centroids at radius 2r (shared by gen_pe and inv)
    int* idx_pe = (int*)aalloc((size_t)B * m * KG * 4);
    {
      int tot = B * m;
      k_ball<<<(tot + 63) / 64, 64, 0, stream>>>(cent, m, cent, m, r2 * r2, idx_pe, tot);
    }

    // inverted residual: xc = relu(bn1(c1 f)); g = max_k(xc[idx]+pe);
    // f' = relu(xc + bn22(c22 relu(bn21(c21 g))))
    float* xc = (float*)aalloc((size_t)B * m * co * 4);
    gemm(fstage, SPm(s, 9), SPm(s, 10), SPm(s, 11), nullptr, xc, B, m, co, co, stream);
    float* gmax = (float*)aalloc((size_t)B * m * co * 4);
    {
      int tot = B * m * co;
      k_groupmax<<<(tot + 255) / 256, 256, 0, stream>>>(
          idx_pe, cent, m, cent, xc, SPm(s, 6), SPm(s, 7), SPm(s, 8), 1.f / r2,
          gmax, m, co, tot);
    }
    float* h1 = (float*)aalloc((size_t)B * m * co * 4);
    gemm(gmax, SPm(s, 12), SPm(s, 13), SPm(s, 14), nullptr, h1, B, m, co, co, stream);
    gemm(h1, SPm(s, 15), SPm(s, 16), SPm(s, 17), xc, fout, B, m, co, co, stream);

    Plv[s + 1] = cent;
    Flv[s + 1] = fout;
    off = mark;  // release transients (stream ordering makes reuse safe)
  }

  // ---- feature propagation (coarse -> fine), levels 3,2,1,0
  const int fpu[4] = {512, 256, 128, 64};
  for (int l = 0; l < 4; ++l) {
    const int lev = 3 - l;
    const int n1 = ns[lev], n2 = ns[lev + 1];
    const int c1 = cs[lev], c2 = cs[lev + 1];
    const int u = fpu[l];  // == c1
    float* fnew = (float*)aalloc((size_t)B * n1 * u * 4);  // persists to next iter
    size_t mark = off;

    int* wi = (int*)aalloc((size_t)B * n1 * 3 * 4);
    float* ww = (float*)aalloc((size_t)B * n1 * 3 * 4);
    {
      int tot = B * n1;
      k_knn3<<<(tot + 255) / 256, 256, 0, stream>>>(Plv[lev], n1, Plv[lev + 1], n2,
                                                    wi, ww, tot);
    }
    const int Ct = c1 + c2;
    float* hcat = (float*)aalloc((size_t)B * n1 * Ct * 4);
    {
      int tot = B * n1 * Ct;
      k_interp_cat<<<(tot + 255) / 256, 256, 0, stream>>>(
          Flv[lev + 1], c2, Flv[lev], c1, wi, ww, hcat, n1, n2, tot);
    }
    float* hmid = (float*)aalloc((size_t)B * n1 * u * 4);
    gemm(hcat, FPm(l, 0), FPm(l, 1), FPm(l, 2), nullptr, hmid, B, n1, Ct, u, stream);
    gemm(hmid, FPm(l, 3), FPm(l, 4), FPm(l, 5), nullptr, fnew, B, n1, u, u, stream);
    Flv[lev] = fnew;
    off = mark;
  }

  // ---- segmentation head
  float* hh = (float*)aalloc((size_t)B * N * 64 * 4);
  gemm(Flv[0], out_c1, out_g, out_b, nullptr, hh, B, N, 64, 64, stream);
  {
    int tot = B * N * 13;
    k_head<<<(tot + 255) / 256, 256, 0, stream>>>(hh, out_c2, out_b2, (float*)d_out, N,
                                                  tot);
  }
}
